// ODEHypernet2D_78872779424230
// MI455X (gfx1250) — matrix operand order repaired and checked by reading, not compiled, not run on hardware
//
#include <hip/hip_runtime.h>

typedef __bf16 v16bf __attribute__((ext_vector_type(16)));
typedef float  v8f   __attribute__((ext_vector_type(8)));
typedef int    v4i   __attribute__((vector_size(4 * sizeof(int))));

#define BATCH   64
#define NPTS    2048
#define PTS     64      // points per block
#define THREADS 128     // 4 wave32s
#define APITCH  304     // bf16 per activation row (K capacity 288 + pad), 608B, 16B aligned
#define WPITCH  40      // bf16 per weight-tile row (K=32 + 8 pad), 80B, 16B aligned
#define TOTAL_W 136716
#define WSB_U16 160256  // prepacked bf16 weight image elements per batch

#define AS1 __attribute__((address_space(1)))
#define AS3 __attribute__((address_space(3)))

#if defined(__AMDGCN__) && __has_builtin(__builtin_amdgcn_global_load_async_to_lds_b128)
#define HAVE_ASYNC_LDS 1
#else
#define HAVE_ASYNC_LDS 0
#endif

// Prepacked image offsets per layer (u16 units, within one batch's WSB_U16 region).
// layer chunk image = COLS x 32 bf16, [col][k] k-fastest (dense, pitch 32).
__device__ __constant__ const int IMG_OFF[4] = {0, 8192, 81920, 155648};
__device__ __constant__ const int L_WOFF[4]  = {0, 2048, 69120, 136192};
__device__ __constant__ const int L_KL[4]    = {4, 258, 258, 258};
__device__ __constant__ const int L_DOUT[4]  = {256, 256, 256, 2};
__device__ __constant__ const int L_COLS[4]  = {256, 256, 256, 16};

struct SMem {
  alignas(16) unsigned short act[PTS][APITCH];  // 38912 B  activations, A-layout [m][k]
  alignas(16) unsigned short wt[256][WPITCH];   // 20480 B  weight chunk, transposed [col][k]
  float ctx[PTS];                               //   256 B
};                                              // 59648 B total -> 5 blocks per 320KB WGP

__device__ __forceinline__ unsigned short f2bf(float f) {
  unsigned int u = __float_as_uint(f);
  u += 0x7FFFu + ((u >> 16) & 1u);              // round-to-nearest-even
  return (unsigned short)(u >> 16);
}
// sigmoid via v_rcp_f32 (avoid IEEE div sequence)
__device__ __forceinline__ float sigmoid_f(float x) {
  return __builtin_amdgcn_rcpf(1.f + __expf(-x));
}
// branch-free stable softplus: max(x,0) + log(1 + exp(-|x|))
__device__ __forceinline__ float softplus_f(float x) {
  const float t = __expf(-__builtin_fabsf(x));
  return fmaxf(x, 0.f) + __logf(1.f + t);
}

#if HAVE_ASYNC_LDS
__device__ __forceinline__ AS3 v4i* lds_cast(void* p) {
  // low 32 bits of a generic shared address == LDS byte offset
  return (AS3 v4i*)(unsigned)(unsigned long long)p;
}
__device__ __forceinline__ AS1 v4i* glob_cast(const void* p) {
  return (AS1 v4i*)(unsigned long long)p;
}
#endif

union FragAB { uint4 u[2]; v16bf v; };

// One hypernet layer. KL = real K (d_in+2), KP = padded K, DOUT = out channels.
// PK: weights come from the prepacked bf16 image (async-copy to LDS).
template <int KL, int KP, int DOUT, bool LAST, bool PK>
__device__ __forceinline__ void run_layer(SMem& s, const float* __restrict__ w,
                                          const unsigned short* __restrict__ wbase,
                                          int woff, int imgoff, int tid,
                                          float* __restrict__ out, int b, int n0) {
  constexpr int NT     = (DOUT + 15) / 16;
  constexpr int COLS   = NT * 16;
  constexpr int NCHUNK = KP / 32;
  const int lane = tid & 31;
  const int wave = tid >> 5;
  const int lo   = lane & 15;
  const int hi   = lane >> 4;

  v8f acc[NT];
#pragma unroll
  for (int nt = 0; nt < NT; ++nt)
#pragma unroll
    for (int r = 0; r < 8; ++r) acc[nt][r] = 0.f;

  const unsigned short* arow = &s.act[wave * 16 + lo][0];

  for (int c = 0; c < NCHUNK; ++c) {
    __syncthreads();
    if constexpr (PK) {
      // Pull bf16 chunk image into bank-padded LDS layout, 16B per lane per op.
      const unsigned short* cimg = wbase + imgoff + c * (COLS * 32);
#if HAVE_ASYNC_LDS
      for (int i = tid; i < COLS * 4; i += THREADS) {
        __builtin_amdgcn_global_load_async_to_lds_b128(
            glob_cast(cimg + i * 8),
            lds_cast(&s.wt[0][0] + (i >> 2) * WPITCH + (i & 3) * 8),
            0, 0);
      }
#if __has_builtin(__builtin_amdgcn_s_wait_asynccnt)
      __builtin_amdgcn_s_wait_asynccnt(0);
#else
      asm volatile("s_wait_asynccnt 0" ::: "memory");
#endif
#else
      for (int i = tid; i < COLS * 4; i += THREADS) {
        const uint4 v = *(const uint4*)(cimg + i * 8);
        *(uint4*)(&s.wt[0][0] + (i >> 2) * WPITCH + (i & 3) * 8) = v;
      }
#endif
    } else {
      // Direct path: coalesced fp32 read -> bf16, transposed into LDS.
      for (int i = tid; i < COLS * 32; i += THREADS) {
        const int col = i % COLS;
        const int k   = i / COLS;
        const int kg  = c * 32 + k;
        float v = 0.f;
        if (kg < KL && col < DOUT) v = w[woff + kg * DOUT + col];
        s.wt[col][k] = f2bf(v);
      }
      if (c + 1 < NCHUNK)
        __builtin_prefetch(&w[woff + (c + 1) * 32 * DOUT + tid * 4], 0, 1);
    }
    __syncthreads();

    // A fragment (16x32 bf16): lanes 0-15 K 0..7 & 16..23; lanes 16-31 K 8..15 & 24..31
    FragAB A;
    A.u[0] = *(const uint4*)(arow + c * 32 + 8 * hi);
    A.u[1] = *(const uint4*)(arow + c * 32 + 16 + 8 * hi);

#pragma unroll
    for (int nt = 0; nt < NT; ++nt) {
      // B fragment (32x16 bf16): lanes 0-15 K 0..7 & 16..23 of col N; lanes 16-31 K 8..15 & 24..31
      FragAB Bf;
      const unsigned short* bcol = &s.wt[nt * 16 + lo][0];
      Bf.u[0] = *(const uint4*)(bcol + 8 * hi);
      Bf.u[1] = *(const uint4*)(bcol + 16 + 8 * hi);
      acc[nt] = __builtin_amdgcn_wmma_f32_16x16x32_bf16(
          false, A.v, false, Bf.v, (short)0, acc[nt], false, false);
    }
  }
  __syncthreads();  // all act reads complete before overwrite

  // Epilogue on C layout: VGPR r -> row r + 8*hi, lane%16 -> column
  const int boff  = woff + KL * DOUT;
  const int scoff = boff + DOUT;
  const int bsoff = scoff + DOUT;
  const int shoff = bsoff + DOUT;
  const int moff  = wave * 16 + 8 * hi;

  float ctxv[8];
#pragma unroll
  for (int r = 0; r < 8; ++r) ctxv[r] = s.ctx[moff + r];

#pragma unroll
  for (int nt = 0; nt < NT; ++nt) {
    const int  ch   = nt * 16 + lo;
    const bool chok = (ch < DOUT);
    const float bias = chok ? w[boff + ch]  : 0.f;
    const float wsc  = chok ? w[scoff + ch] : 0.f;
    const float bsc  = chok ? w[bsoff + ch] : 0.f;
    const float wsh  = chok ? w[shoff + ch] : 0.f;
#pragma unroll
    for (int r = 0; r < 8; ++r) {
      float x = acc[nt][r] + bias;
      const float sg = sigmoid_f(ctxv[r] * wsc + bsc);
      x = sg * x + ctxv[r] * wsh;
      if (!LAST) {
        x = softplus_f(x);
        s.act[moff + r][ch] = f2bf(x);
      } else if (ch < 2) {
        out[((size_t)(b * NPTS + n0 + moff + r)) * 2 + ch] = x;
      }
    }
  }
  __syncthreads();
}

// Prepack: fp32 weights -> bf16, transposed [col][k] per 32-row chunk, padded with zeros.
__global__ void __launch_bounds__(256)
prep_weights(const float* __restrict__ tnw, unsigned short* __restrict__ img) {
  __shared__ float tile[32][257];
  const int b = blockIdx.y;
  const int x = blockIdx.x;  // 0..27 -> (layer, chunk)
  int l, c;
  if (x == 0)      { l = 0; c = 0;      }
  else if (x < 10) { l = 1; c = x - 1;  }
  else if (x < 19) { l = 2; c = x - 10; }
  else             { l = 3; c = x - 19; }
  const int KL = L_KL[l], DOUT = L_DOUT[l], COLS = L_COLS[l];
  const float* w = tnw + (size_t)b * TOTAL_W + L_WOFF[l];
  unsigned short* o = img + (size_t)b * WSB_U16 + IMG_OFF[l] + c * COLS * 32;
  const int tid = threadIdx.x;

  for (int i = tid; i < 32 * COLS; i += 256) {   // coalesced fp32 reads
    const int col = i % COLS, k = i / COLS;
    const int kg  = c * 32 + k;
    tile[k][col] = (kg < KL && col < DOUT) ? w[kg * DOUT + col] : 0.f;
  }
  __syncthreads();
  for (int i = tid; i < COLS * 32; i += 256) {   // coalesced bf16 writes, transposed
    const int col = i >> 5, k = i & 31;
    o[i] = f2bf(tile[k][col]);
  }
}

template <bool PK>
__global__ void __launch_bounds__(THREADS)
hypernet2d_kernel(const float* __restrict__ context, const float* __restrict__ y,
                  const float* __restrict__ y_points, const float* __restrict__ tnw,
                  const unsigned short* __restrict__ wsimg, float* __restrict__ out) {
  __shared__ SMem s;
  const int tile = blockIdx.x;     // 0..31 point tile
  const int b    = blockIdx.y;     // 0..63 batch
  const int n0   = tile * PTS;
  const int tid  = threadIdx.x;
  const float* w = tnw + (size_t)b * TOTAL_W;
  const unsigned short* wbase = PK ? (wsimg + (size_t)b * WSB_U16) : nullptr;

  // Zero activation buffer (covers all K padding columns once).
  {
    uint4 z; z.x = 0u; z.y = 0u; z.z = 0u; z.w = 0u;
    uint4* p = (uint4*)&s.act[0][0];
    for (int i = tid; i < PTS * APITCH / 8; i += THREADS) p[i] = z;
  }
  __syncthreads();

  // Init: context; layer-0 inputs [y, y_points] in cols 0..3; y_points pinned at 256..257.
  for (int i = tid; i < PTS; i += THREADS) {
    const size_t g = (size_t)b * NPTS + n0 + i;
    s.ctx[i] = context[g];
    const float2 yv = ((const float2*)y)[g];
    const float2 yp = ((const float2*)y_points)[g];
    s.act[i][0]   = f2bf(yv.x);
    s.act[i][1]   = f2bf(yv.y);
    s.act[i][2]   = f2bf(yp.x);
    s.act[i][3]   = f2bf(yp.y);
    s.act[i][256] = f2bf(yp.x);
    s.act[i][257] = f2bf(yp.y);
  }
  __syncthreads();

  run_layer<  4,  32, 256, false, PK>(s, w, wbase,      0,      0, tid, out, b, n0);
  run_layer<258, 288, 256, false, PK>(s, w, wbase,   2048,   8192, tid, out, b, n0);
  run_layer<258, 288, 256, false, PK>(s, w, wbase,  69120,  81920, tid, out, b, n0);
  run_layer<258, 288,   2, true , PK>(s, w, wbase, 136192, 155648, tid, out, b, n0);
}

extern "C" void kernel_launch(void* const* d_in, const int* in_sizes, int n_in,
                              void* d_out, int out_size, void* d_ws, size_t ws_size,
                              hipStream_t stream) {
  const float* context  = (const float*)d_in[0];
  const float* y        = (const float*)d_in[1];
  const float* y_points = (const float*)d_in[2];
  const float* tnw      = (const float*)d_in[3];
  float* out = (float*)d_out;

  const size_t need = (size_t)BATCH * WSB_U16 * 2;  // ~19.6 MB bf16 weight image
  dim3 grid(NPTS / PTS, BATCH);                     // 32 x 64 blocks

  if (d_ws != nullptr && ws_size >= need) {
    prep_weights<<<dim3(28, BATCH), 256, 0, stream>>>(tnw, (unsigned short*)d_ws);
    hypernet2d_kernel<true><<<grid, THREADS, 0, stream>>>(
        context, y, y_points, tnw, (const unsigned short*)d_ws, out);
  } else {
    hypernet2d_kernel<false><<<grid, THREADS, 0, stream>>>(
        context, y, y_points, tnw, nullptr, out);
  }
}
